// NST_249108103833
// MI455X (gfx1250) — matrix operand order repaired and checked by reading, not compiled, not run on hardware
//
#include <hip/hip_runtime.h>
#include <hip/hip_bf16.h>

#define BATCH 64
#define CH    512
#define LEN   784
#define KPAD  800            // LEN zero-padded to multiple of 32
#define KCHUNKS 25           // KPAD / 32
#define LDSPITCH 40          // 32 k-values + 8 pad (ushorts) per row
#define STAGEUS (64 * LDSPITCH)   // ushorts per LDS stage per array

typedef __attribute__((ext_vector_type(16))) __bf16 v16bf;
typedef __attribute__((ext_vector_type(8)))  float  v8f;

union Frag {
    v16bf bf;
    uint4 q[2];
    unsigned short us[16];
};

// Round-to-nearest-even fp32 -> bf16 hi, plus bf16 residual lo.
__device__ __forceinline__ void split_bf16(float x, unsigned short& hi, unsigned short& lo) {
    unsigned u = __float_as_uint(x);
    unsigned r = u + 0x7FFFu + ((u >> 16) & 1u);
    hi = (unsigned short)(r >> 16);
    float hf = __uint_as_float(((unsigned)hi) << 16);
    float l = x - hf;
    unsigned ul = __float_as_uint(l);
    unsigned rl = ul + 0x7FFFu + ((ul >> 16) & 1u);
    lo = (unsigned short)(rl >> 16);
}

// One wave per row: L2-normalize, split to bf16 hi/lo planes, zero-pad K to KPAD.
__global__ __launch_bounds__(256) void norm_split(const float* __restrict__ x,
                                                  unsigned short* __restrict__ hi_plane,
                                                  unsigned short* __restrict__ lo_plane,
                                                  int nrows) {
    int wave = threadIdx.x >> 5;
    int lane = threadIdx.x & 31;
    int row  = blockIdx.x * 8 + wave;
    if (row >= nrows) return;
    const float* p = x + (size_t)row * LEN;

    float s = 0.0f;
    for (int l = lane; l < LEN; l += 32) {
        float v = p[l];
        s += v * v;
    }
    #pragma unroll
    for (int off = 16; off > 0; off >>= 1)
        s += __shfl_xor(s, off, 32);
    float inv = 1.0f / fmaxf(sqrtf(s), 1e-12f);   // all lanes hold total

    unsigned short* ph = hi_plane + (size_t)row * KPAD;
    unsigned short* pl = lo_plane + (size_t)row * KPAD;
    for (int l = lane; l < KPAD; l += 32) {
        float v = (l < LEN) ? p[l] * inv : 0.0f;
        unsigned short h, lo;
        split_bf16(v, h, lo);
        ph[l] = h;
        pl[l] = lo;
    }
}

// One WG per (gram, batch, 64x64 tile), upper-triangular tiles only for the
// two symmetric grams (off-diagonal weight 2). 8 waves; each wave owns two
// 16x16 output tiles. Double-buffered LDS: one barrier per 32-K chunk, global
// prefetch issued inside the compute region so its latency hides under WMMA.
#define TILES_PER_BATCH 136   // 36 (tt upper) + 36 (ss upper) + 64 (st full)

__global__ __launch_bounds__(256) void gram_tiles(const unsigned short* __restrict__ s_hi,
                                                  const unsigned short* __restrict__ s_lo,
                                                  const unsigned short* __restrict__ t_hi,
                                                  const unsigned short* __restrict__ t_lo,
                                                  float* __restrict__ partials) {
    __shared__ unsigned short Ahi[2 * STAGEUS];
    __shared__ unsigned short Alo[2 * STAGEUS];
    __shared__ unsigned short Bhi[2 * STAGEUS];
    __shared__ unsigned short Blo[2 * STAGEUS];
    __shared__ float redbuf[8];

    const int wg = blockIdx.x;
    const int u  = wg % TILES_PER_BATCH;
    const int b  = wg / TILES_PER_BATCH;

    int ti, tj;
    float w;
    const unsigned short *Xhi, *Xlo, *Yhi, *Ylo;
    if (u < 72) {
        // symmetric grams: decode upper-triangular 8x8 tile index
        const int g = (u < 36) ? 0 : 1;
        int rem = u - g * 36;
        int r = 0;
        while (rem >= 8 - r) { rem -= 8 - r; ++r; }
        ti = r;
        tj = r + rem;
        w = (ti == tj) ? 1.0f : 2.0f;
        if (g == 0) { Xhi = t_hi; Xlo = t_lo; Yhi = t_hi; Ylo = t_lo; }
        else        { Xhi = s_hi; Xlo = s_lo; Yhi = s_hi; Ylo = s_lo; }
    } else {
        const int v = u - 72;
        ti = v >> 3;
        tj = v & 7;
        w  = -2.0f;
        Xhi = s_hi; Xlo = s_lo; Yhi = t_hi; Ylo = t_lo;
    }
    const int i0 = ti * 64;
    const int j0 = tj * 64;

    const int tid  = threadIdx.x;
    const int wave = tid >> 5;
    const int lane = tid & 31;

    // wave -> output sub-tiles: 4 row-groups x 2 col-groups (of 32 cols)
    const int mt = (wave >> 1) * 16;
    const int ct = (wave & 1) * 32;

    v8f acc0 = {};
    v8f acc1 = {};

    // loader mapping: thread -> (row 0..63, quarter q 0..3 of the 32-wide chunk)
    const int lrow = tid >> 2;
    const int lq   = tid & 3;
    const size_t rowA = ((size_t)b * CH + i0 + lrow) * KPAD + lq * 8;
    const size_t rowB = ((size_t)b * CH + j0 + lrow) * KPAD + lq * 8;
    const int lds_idx = lrow * LDSPITCH + lq * 8;   // 16B aligned

    // fragment gather mapping (wave32): contiguous k = 16*(lane/16) + s, same for A and B
    const int fm    = lane & 15;
    const int kqA   = (lane >> 4) * 2;              // A: {0,2} -> uint4 offset
    const int kqB   = (lane >> 4) * 2;              // B: {0,2} -> uint4 offset
    const int arow5  = (mt + fm) * (LDSPITCH / 8);
    const int b0row5 = (ct + fm) * (LDSPITCH / 8);
    const int b1row5 = (ct + 16 + fm) * (LDSPITCH / 8);
    const uint4* Ahi4 = (const uint4*)Ahi;
    const uint4* Alo4 = (const uint4*)Alo;
    const uint4* Bhi4 = (const uint4*)Bhi;
    const uint4* Blo4 = (const uint4*)Blo;

    // ---- prologue: buf0 <- chunk 0, regs <- chunk 1 ----
    uint4 rah = *(const uint4*)(Xhi + rowA);
    uint4 ral = *(const uint4*)(Xlo + rowA);
    uint4 rbh = *(const uint4*)(Yhi + rowB);
    uint4 rbl = *(const uint4*)(Ylo + rowB);
    *(uint4*)&Ahi[lds_idx] = rah;
    *(uint4*)&Alo[lds_idx] = ral;
    *(uint4*)&Bhi[lds_idx] = rbh;
    *(uint4*)&Blo[lds_idx] = rbl;
    rah = *(const uint4*)(Xhi + rowA + 32);
    ral = *(const uint4*)(Xlo + rowA + 32);
    rbh = *(const uint4*)(Yhi + rowB + 32);
    rbl = *(const uint4*)(Ylo + rowB + 32);
    __syncthreads();

    // invariant at top of iter kc: buf[p] = chunk kc, regs = chunk kc+1
    for (int kc = 0; kc < KCHUNKS; ++kc) {
        const int p  = kc & 1;
        const int sp = p * STAGEUS;          // compute stage offset (ushorts)
        const int sq = (p ^ 1) * STAGEUS;    // fill stage offset
        const int s5 = p * (STAGEUS / 8);    // compute stage offset (uint4)

        // store chunk kc+1 into the other buffer (prev barrier => safe)
        *(uint4*)&Ahi[sq + lds_idx] = rah;
        *(uint4*)&Alo[sq + lds_idx] = ral;
        *(uint4*)&Bhi[sq + lds_idx] = rbh;
        *(uint4*)&Blo[sq + lds_idx] = rbl;

        // fragment loads for chunk kc
        Frag ah, al, bh0, bl0, bh1, bl1;
        ah.q[0] = Ahi4[s5 + arow5 + kqA];
        ah.q[1] = Ahi4[s5 + arow5 + kqA + 1];
        al.q[0] = Alo4[s5 + arow5 + kqA];
        al.q[1] = Alo4[s5 + arow5 + kqA + 1];
        bh0.q[0] = Bhi4[s5 + b0row5 + kqB];
        bh0.q[1] = Bhi4[s5 + b0row5 + kqB + 1];
        bl0.q[0] = Blo4[s5 + b0row5 + kqB];
        bl0.q[1] = Blo4[s5 + b0row5 + kqB + 1];
        bh1.q[0] = Bhi4[s5 + b1row5 + kqB];
        bh1.q[1] = Bhi4[s5 + b1row5 + kqB + 1];
        bl1.q[0] = Blo4[s5 + b1row5 + kqB];
        bl1.q[1] = Blo4[s5 + b1row5 + kqB + 1];

        // global prefetch of chunk kc+2 (clamped: harmless tail reload),
        // issued in the compute region so latency hides under WMMA
        {
            const int kn = (kc + 2 < KCHUNKS) ? (kc + 2) : (KCHUNKS - 1);
            const size_t offn = (size_t)kn * 32;
            rah = *(const uint4*)(Xhi + rowA + offn);
            ral = *(const uint4*)(Xlo + rowA + offn);
            rbh = *(const uint4*)(Yhi + rowB + offn);
            rbl = *(const uint4*)(Ylo + rowB + offn);
        }

        // error-compensated product: hi*hi + hi*lo + lo*hi  (drop lo*lo)
        acc0 = __builtin_amdgcn_wmma_f32_16x16x32_bf16(false, ah.bf, false, bh0.bf, (short)0, acc0, false, false);
        acc0 = __builtin_amdgcn_wmma_f32_16x16x32_bf16(false, ah.bf, false, bl0.bf, (short)0, acc0, false, false);
        acc0 = __builtin_amdgcn_wmma_f32_16x16x32_bf16(false, al.bf, false, bh0.bf, (short)0, acc0, false, false);
        acc1 = __builtin_amdgcn_wmma_f32_16x16x32_bf16(false, ah.bf, false, bh1.bf, (short)0, acc1, false, false);
        acc1 = __builtin_amdgcn_wmma_f32_16x16x32_bf16(false, ah.bf, false, bl1.bf, (short)0, acc1, false, false);
        acc1 = __builtin_amdgcn_wmma_f32_16x16x32_bf16(false, al.bf, false, bh1.bf, (short)0, acc1, false, false);

        // single barrier per chunk: my fill-stores done (visible next iter),
        // all waves' reads of buf[p] done (safe to overwrite next iter)
        __syncthreads();
    }

    // sum of squares of this wave's gram entries
    float s = 0.0f;
    #pragma unroll
    for (int i = 0; i < 8; ++i) {
        s += acc0[i] * acc0[i];
        s += acc1[i] * acc1[i];
    }
    #pragma unroll
    for (int off = 16; off > 0; off >>= 1)
        s += __shfl_xor(s, off, 32);
    if (lane == 0) redbuf[wave] = s;
    __syncthreads();
    if (tid == 0) {
        float tot = 0.0f;
        #pragma unroll
        for (int i = 0; i < 8; ++i) tot += redbuf[i];
        partials[wg] = w * tot;
    }
}

__global__ __launch_bounds__(256) void final_reduce(const float* __restrict__ partials,
                                                    int n, float* __restrict__ out) {
    __shared__ float sb[256];
    float s = 0.0f;
    for (int i = threadIdx.x; i < n; i += 256) s += partials[i];
    sb[threadIdx.x] = s;
    __syncthreads();
    for (int off = 128; off > 0; off >>= 1) {
        if (threadIdx.x < off) sb[threadIdx.x] += sb[threadIdx.x + off];
        __syncthreads();
    }
    if (threadIdx.x == 0)
        out[0] = sb[0] / ((float)BATCH * (float)CH * (float)CH);
}

extern "C" void kernel_launch(void* const* d_in, const int* in_sizes, int n_in,
                              void* d_out, int out_size, void* d_ws, size_t ws_size,
                              hipStream_t stream) {
    const float* fs = (const float*)d_in[0];
    const float* ft = (const float*)d_in[1];

    const size_t plane_elems = (size_t)BATCH * CH * KPAD;   // 26,214,400 ushorts
    unsigned short* s_hi = (unsigned short*)d_ws;
    unsigned short* s_lo = s_hi + plane_elems;
    unsigned short* t_hi = s_lo + plane_elems;
    unsigned short* t_lo = t_hi + plane_elems;
    float* partials = (float*)(t_lo + plane_elems);         // 8704 floats

    const int nrows = BATCH * CH;                           // 32768
    norm_split<<<nrows / 8, 256, 0, stream>>>(fs, s_hi, s_lo, nrows);
    norm_split<<<nrows / 8, 256, 0, stream>>>(ft, t_hi, t_lo, nrows);

    const int nwg = BATCH * TILES_PER_BATCH;                // 8704
    gram_tiles<<<nwg, 256, 0, stream>>>(s_hi, s_lo, t_hi, t_lo, partials);

    final_reduce<<<1, 256, 0, stream>>>(partials, nwg, (float*)d_out);
}
